// Sparse_Kernel_Matrix_Multiplication_65747359367885
// MI455X (gfx1250) — matrix operand order repaired and checked by reading, not compiled, not run on hardware
//
#include <hip/hip_runtime.h>
#include <hip/hip_bf16.h>

typedef __attribute__((ext_vector_type(16))) _Float16 v16h;
typedef __attribute__((ext_vector_type(8)))  float    v8f;
typedef __attribute__((ext_vector_type(8)))  unsigned int v8u;

#define BB 2
#define HH 128
#define WW 128
#define CC 16
#define FF 16
// px=py=8, sx=sy=4 -> patch grid 32x32 (SAME pad 2), tile grid 33x33 of 4x4 pixels

// ---------------- Phase 1: per-patch kernel-weighted normalization ----------
// ratio[b,p,q,c,f] = divide_no_nan( sum(pet*b), sum(b) ) over the 8x8 patch.
__global__ __launch_bounds__(256) void ratio_kernel(
    const float* __restrict__ pet, const float* __restrict__ bfeat,
    float* __restrict__ ratio) {
  const int blk = blockIdx.x;
  const int q = blk & 31, p = (blk >> 5) & 31, b = blk >> 10;
  const int tid = threadIdx.x;
  const int c = tid >> 4, f = tid & 15;

  __shared__ float petS[64 * CC];           // 8x8 patch of pet, all channels
  for (int i = tid; i < 64 * CC; i += 256) {
    int pix = i >> 4, cc = i & 15;
    int u = pix >> 3, v = pix & 7;
    int r = 4 * p - 2 + u, s = 4 * q - 2 + v;
    float val = 0.f;
    if ((unsigned)r < HH && (unsigned)s < WW)   // scalar-uniform per block row/col
      val = pet[(((size_t)b * HH + r) * WW + s) * CC + cc];
    petS[i] = val;
  }
  __syncthreads();

  float diag = 0.f, kp = 0.f;
  const float* bb = bfeat + (size_t)b * HH * WW * CC * FF + c * FF + f;
  #pragma unroll
  for (int u = 0; u < 8; ++u) {
    int r = 4 * p - 2 + u;                   // uniform per block -> scalar branch
    if ((unsigned)r >= HH) continue;
    const float* brow = bb + (size_t)r * WW * CC * FF;
    __builtin_prefetch(brow, 0, 1);          // -> global_prefetch_b8
    #pragma unroll
    for (int v = 0; v < 8; ++v) {
      int s = 4 * q - 2 + v;                 // uniform per block -> scalar branch
      if ((unsigned)s >= WW) continue;
      float bv = brow[(size_t)s * CC * FF];
      float pv = petS[(u * 8 + v) * CC + c];
      diag += bv;
      kp = fmaf(pv, bv, kp);
    }
  }
  float outv = (diag == 0.f) ? 0.f : kp / diag;
  ratio[((((size_t)b * 32 + p) * 32 + q) * CC + c) * FF + f] = outv;
}

// ---------------- Phase 2: WMMA fusion over 4x4 output tiles ----------------
// One wave per tile T,S in [0,33): pixels y=4T-2+(m>>2), x=4S-2+(m&3), m=0..15.
// D[16pix x 16c] = A[16pix x 256(cf)] * blockdiag(0.25*Rsum)[256 x 16c],
// accumulated over 8 chunks of K=32 with v_wmma_f32_16x16x32_f16.
// Out-of-image A rows load clamped (valid) addresses unconditionally; their D
// rows are garbage but never stored, so no per-element predication is needed.
__global__ __launch_bounds__(128) void fuse_kernel(
    const float* __restrict__ bfeat, const float* __restrict__ ratio,
    float* __restrict__ out) {
  const int NT = BB * 33 * 33;
  const int wave = blockIdx.x * (blockDim.x >> 5) + (threadIdx.x >> 5);
  if (wave >= NT) return;                    // wave-uniform: EXEC stays all-ones
  const int lane = threadIdx.x & 31;
  const bool lo = lane < 16;
  const int cN = lane & 15;                  // B/D column (channel)

  int t = wave;
  const int S = t % 33; t /= 33;
  const int T = t % 33; t /= 33;
  const int b = t;

  // even/odd patch pair covering this tile in each axis; weight 0 if invalid
  const int  p0 = 2 * (T >> 1);
  const int  p1 = 2 * ((T - 1) >> 1) + 1;    // T=0 -> -1 (invalid)
  const int  q0 = 2 * (S >> 1);
  const int  q1 = 2 * ((S - 1) >> 1) + 1;
  const bool p0v = (T < 32), p1v = (T > 0);
  const bool q0v = (S < 32), q1v = (S > 0);

  // Rsum row for channel cN: 0.25 * sum of up to 4 ratio vectors, branchless.
  float acc[16];
  #pragma unroll
  for (int i = 0; i < 16; ++i) acc[i] = 0.f;
  const int   pqi[4][2] = {{p0, q0}, {p0, q1}, {p1, q0}, {p1, q1}};
  const float pqw[4] = {(p0v && q0v) ? 0.25f : 0.f, (p0v && q1v) ? 0.25f : 0.f,
                        (p1v && q0v) ? 0.25f : 0.f, (p1v && q1v) ? 0.25f : 0.f};
  #pragma unroll
  for (int j = 0; j < 4; ++j) {
    int p_ = pqi[j][0] & 31;                 // clamp (-1 -> harmless valid idx)
    int q_ = pqi[j][1] & 31;
    float w = pqw[j];
    const float4* rp = (const float4*)(ratio +
        ((((size_t)b * 32 + p_) * 32 + q_) * CC + cN) * FF);
    #pragma unroll
    for (int m4 = 0; m4 < 4; ++m4) {
      float4 v = rp[m4];                     // unconditional load
      acc[4 * m4 + 0] = fmaf(w, v.x, acc[4 * m4 + 0]);
      acc[4 * m4 + 1] = fmaf(w, v.y, acc[4 * m4 + 1]);
      acc[4 * m4 + 2] = fmaf(w, v.z, acc[4 * m4 + 2]);
      acc[4 * m4 + 3] = fmaf(w, v.w, acc[4 * m4 + 3]);
    }
  }
  v16h rsum;
  #pragma unroll
  for (int i = 0; i < 16; ++i) rsum[i] = (_Float16)acc[i];
  v8u rsum32 = *(const v8u*)&rsum;           // 32-bit view for whole-vector select

  // A-matrix pixel for this lane (A layout: lanes 0-15/16-31 both map M=lane&15)
  const int m = lane & 15;
  int y = 4 * T - 2 + (m >> 2);
  int x = 4 * S - 2 + (m & 3);
  y = min(max(y, 0), HH - 1);                // clamp: always-valid address
  x = min(max(x, 0), WW - 1);
  const float4* pp = (const float4*)(bfeat +
      (((size_t)b * HH + y) * WW + x) * (CC * FF));
  const int f04 = lo ? 0 : 4;                // c-offset 0 or 1 (4 float4 per channel)

  v8f cacc = {};
  #pragma unroll
  for (int k = 0; k < 8; ++k) {
    // A chunk k: halves[0..15] = bfeat[pix, c=2k+h, f0..f15]   (h = lo?0:1)
    float4 a0 = pp[8 * k + f04];             // unconditional b128 loads
    float4 a1 = pp[8 * k + f04 + 1];
    float4 a2 = pp[8 * k + f04 + 2];
    float4 a3 = pp[8 * k + f04 + 3];
    v16h a;
    a[0]  = (_Float16)a0.x; a[1]  = (_Float16)a0.y; a[2]  = (_Float16)a0.z; a[3]  = (_Float16)a0.w;
    a[4]  = (_Float16)a1.x; a[5]  = (_Float16)a1.y; a[6]  = (_Float16)a1.z; a[7]  = (_Float16)a1.w;
    a[8]  = (_Float16)a2.x; a[9]  = (_Float16)a2.y; a[10] = (_Float16)a2.z; a[11] = (_Float16)a2.w;
    a[12] = (_Float16)a3.x; a[13] = (_Float16)a3.y; a[14] = (_Float16)a3.z; a[15] = (_Float16)a3.w;

    // B chunk k (block-diagonal): lane carries column c'=cN; halves are
    // K = 0..15 (lanes 0-15) or 16..31 (lanes 16-31); nonzero iff c' == 2k (+1).
    bool hasB = lo ? (cN == 2 * k) : (cN == 2 * k + 1);
    v8u bu = hasB ? rsum32 : (v8u)0;         // 8x v_cndmask_b32
    v16h bm = *(const v16h*)&bu;

    cacc = __builtin_amdgcn_wmma_f32_16x16x32_f16(
        false, a, false, bm, (short)0, cacc, false, false);
  }

  // D layout: lane = column cN; VGPR r holds row M = r + (lo ? 0 : 8)
  #pragma unroll
  for (int r = 0; r < 8; ++r) {
    int M  = r + (lo ? 0 : 8);
    int yy = 4 * T - 2 + (M >> 2);
    int xx = 4 * S - 2 + (M & 3);
    if ((unsigned)yy < HH && (unsigned)xx < WW)
      out[(((size_t)b * HH + yy) * WW + xx) * CC + cN] = cacc[r];
  }
}

extern "C" void kernel_launch(void* const* d_in, const int* in_sizes, int n_in,
                              void* d_out, int out_size, void* d_ws, size_t ws_size,
                              hipStream_t stream) {
  // inputs: 0=mr (unused), 1=pet (B,H,W,C) f32, 2=b_features (B,H,W,C,F) f32,
  //         3..6 = px,py,sx,sy scalars (hardcoded: 8,8,4,4)
  const float* pet   = (const float*)d_in[1];
  const float* bfeat = (const float*)d_in[2];
  float* ratio = (float*)d_ws;               // BB*32*32*CC*FF*4 = 2 MB scratch

  ratio_kernel<<<BB * 32 * 32, 256, 0, stream>>>(pet, bfeat, ratio);

  const int NT = BB * 33 * 33;
  const int blocks = (NT + 3) / 4;           // 4 waves (tiles) per 128-thr block
  fuse_kernel<<<blocks, 128, 0, stream>>>(bfeat, ratio, (float*)d_out);
}
